// MicroGNN_14061722927669
// MI455X (gfx1250) — matrix operand order repaired and checked by reading, not compiled, not run on hardware
//
#include <hip/hip_runtime.h>

// ---------------------------------------------------------------------------
// MicroGNN on MI455X (gfx1250):
//   * dense 128-K GEMMs via v_wmma_f32_16x16x32_bf16 (bf16 in, f32 accumulate),
//     templated on NCOLS so the N-tile loop fully unrolls (32 WMMA / wave-slab)
//   * f32->bf16 via vector fptrunc -> v_cvt_pk_bf16_f32 when available
//   * graph propagation (gather * norm -> scatter-add) via f32 global atomics;
//     h2 slab (51.2 MB) fits in the 192 MB L2, so src-gathers / dst-atomics
//     are L2-resident: this is the bandwidth-dominant stage (~8 GB total).
// ---------------------------------------------------------------------------

typedef __attribute__((ext_vector_type(16))) __bf16 v16bf;
typedef __attribute__((ext_vector_type(2)))  __bf16 v2bf;
typedef __attribute__((ext_vector_type(2)))  float  v2f;
typedef __attribute__((ext_vector_type(8)))  float  v8f;

union ABfrag { v16bf v; unsigned int u[8]; };

__device__ __forceinline__ unsigned int f2bf_pk(float lo, float hi) {
  // Vector fptrunc: ISel emits v_cvt_pk_bf16_f32 on targets with bf16 cvt
  // hardware (gfx1250 per FeatureBF16ConversionInsts); else RNE expansion.
  v2f f2; f2.x = lo; f2.y = hi;
  v2bf b2 = __builtin_convertvector(f2, v2bf);
  return __builtin_bit_cast(unsigned int, b2);
}

// ---------------------------------------------------------------------------
// GEMM:  Y[nrows x NCOLS] = X[nrows x 128] * W[128 x NCOLS] (+ bias)
// 256 threads = 8 waves, 128 rows per block. A fragments read straight from
// global (L2-resident) as float4 runs matching the ISA 16-bit A 16x32 layout;
// W^T staged once per block in LDS as bf16 pairs (stride 66 dwords).
// ---------------------------------------------------------------------------
template <int NCOLS>
__global__ __launch_bounds__(256)
void gemm_bf16_wmma(const float* __restrict__ X, const float* __restrict__ W,
                    const float* __restrict__ bias, float* __restrict__ Y,
                    int nrows) {
  __shared__ unsigned int Wst[128 * 66];       // W transposed: [col][k-pair]

  const int tid  = threadIdx.x;
  const int lane = tid & 31;
  const int wave = tid >> 5;
  const int rowbase = blockIdx.x * 128 + wave * 16;

  // --- stage W^T as bf16 pairs (coalesced reads; W is 64 KB, L2-resident) --
  {
    const int col = tid & 127;
    const int kb  = (tid >> 7) * 64;           // threads split K in halves
    for (int j = 0; j < 64; j += 2) {
      int k = kb + j;
      float lo = (col < NCOLS) ? W[(size_t)k * NCOLS + col] : 0.0f;
      float hi = (col < NCOLS) ? W[(size_t)(k + 1) * NCOLS + col] : 0.0f;
      Wst[col * 66 + (k >> 1)] = f2bf_pk(lo, hi);
    }
  }

  const int half = lane >> 4;
  const int m    = lane & 15;

  // --- A fragments straight from global: ISA A-layout K-pair runs are
  //     32B-aligned blocks of 8 floats -> 4x float4 per K-chunk per lane ----
  ABfrag afrag[4];
  {
    int row  = rowbase + m;
    int rowc = row < nrows ? row : (nrows - 1);     // clamp; stores guarded
    const float* xr = X + (size_t)rowc * 128;
    #pragma unroll
    for (int c = 0; c < 4; ++c) {
      const float4 q0 = *(const float4*)(xr + c * 32 + 8 * half);
      const float4 q1 = *(const float4*)(xr + c * 32 + 8 * half + 4);
      const float4 q2 = *(const float4*)(xr + c * 32 + 16 + 8 * half);
      const float4 q3 = *(const float4*)(xr + c * 32 + 16 + 8 * half + 4);
      afrag[c].u[0] = f2bf_pk(q0.x, q0.y); afrag[c].u[1] = f2bf_pk(q0.z, q0.w);
      afrag[c].u[2] = f2bf_pk(q1.x, q1.y); afrag[c].u[3] = f2bf_pk(q1.z, q1.w);
      afrag[c].u[4] = f2bf_pk(q2.x, q2.y); afrag[c].u[5] = f2bf_pk(q2.z, q2.w);
      afrag[c].u[6] = f2bf_pk(q3.x, q3.y); afrag[c].u[7] = f2bf_pk(q3.z, q3.w);
    }
  }
  __syncthreads();

  const bool fullrows = (rowbase + 16 <= nrows);    // wave-uniform fast path
  constexpr int NT = (NCOLS + 15) / 16;

  #pragma unroll
  for (int t = 0; t < NT; ++t) {
    const int ncol = t * 16 + m;                    // this lane's output col
    const unsigned int* brow = &Wst[ncol * 66];
    v8f acc = {};
    #pragma unroll
    for (int c = 0; c < 4; ++c) {
      ABfrag b;
      #pragma unroll
      for (int v = 0; v < 8; ++v) {                 // ISA 16-bit B 32x16 layout
        int klo = c * 32 + half * 8 + 2 * v + (v >= 4 ? 8 : 0);
        b.u[v] = brow[klo >> 1];
      }
      acc = __builtin_amdgcn_wmma_f32_16x16x32_bf16(
                false, afrag[c].v, false, b.v, (short)0, acc, false, false);
    }
    const bool colok = ((t + 1) * 16 <= NCOLS) || (ncol < NCOLS);
    const float bn = (colok && bias) ? bias[ncol] : 0.0f;
    if (fullrows) {
      if (colok) {
        #pragma unroll
        for (int r = 0; r < 8; ++r)                 // D: M = r + 8*half
          Y[(size_t)(rowbase + r + 8 * half) * NCOLS + ncol] = acc[r] + bn;
      }
    } else {
      #pragma unroll
      for (int r = 0; r < 8; ++r) {
        int row = rowbase + r + 8 * half;
        if (colok && row < nrows)
          Y[(size_t)row * NCOLS + ncol] = acc[r] + bn;
      }
    }
  }
}

// ---------------------------------------------------------------------------
// Degree / normalization
// ---------------------------------------------------------------------------
__global__ void deg_init(float* __restrict__ deg, int n) {
  int i = blockIdx.x * blockDim.x + threadIdx.x;
  if (i < n) deg[i] = 1.0f;                    // self-loop
}
__global__ void deg_edges(const int* __restrict__ ei, float* __restrict__ deg,
                          int ne) {
  int e = blockIdx.x * blockDim.x + threadIdx.x;
  if (e < ne) unsafeAtomicAdd(&deg[ei[ne + e]], 1.0f);   // dst = row 1
}
__global__ void deg_rsqrt(float* __restrict__ deg, int n) {
  int i = blockIdx.x * blockDim.x + threadIdx.x;
  if (i < n) deg[i] = rsqrtf(deg[i]);
}

// ---------------------------------------------------------------------------
// Aggregation: out[i] = bias + dinv[i]^2 * h2[i]  (self-loop + bias fold)
// then edge scatter: out[dst] += dinv[src]*dinv[dst] * h2[src]
// ---------------------------------------------------------------------------
__global__ void agg_init(const float* __restrict__ h2,
                         const float* __restrict__ dinv,
                         const float* __restrict__ bias,
                         float* __restrict__ out, int n) {
  int idx = blockIdx.x * blockDim.x + threadIdx.x;   // n * 32 threads
  int node = idx >> 5;
  int c4 = idx & 31;
  if (node >= n) return;
  float dn = dinv[node];
  float s = dn * dn;
  float4 h = ((const float4*)(h2 + (size_t)node * 128))[c4];
  float4 b = ((const float4*)bias)[c4];
  float4 o = { h.x * s + b.x, h.y * s + b.y, h.z * s + b.z, h.w * s + b.w };
  ((float4*)(out + (size_t)node * 128))[c4] = o;
}

__global__ void agg_edges(const float* __restrict__ h2,
                          const int* __restrict__ ei,
                          const float* __restrict__ dinv,
                          float* __restrict__ out, int ne) {
  int gw   = (blockIdx.x * blockDim.x + threadIdx.x) >> 5;  // one wave / edge
  int lane = threadIdx.x & 31;
  if (gw >= ne) return;
  int s = ei[gw];
  int d = ei[ne + gw];
  float nrm = dinv[s] * dinv[d];
  float4 h = ((const float4*)(h2 + (size_t)s * 128))[lane];
  float* op = out + (size_t)d * 128 + lane * 4;
  unsafeAtomicAdd(op + 0, h.x * nrm);
  unsafeAtomicAdd(op + 1, h.y * nrm);
  unsafeAtomicAdd(op + 2, h.z * nrm);
  unsafeAtomicAdd(op + 3, h.w * nrm);
}

// out = 2*g0 + g1  ('add' merge double-counts g0)
__global__ void combine2(const float* __restrict__ g0,
                         const float* __restrict__ g1,
                         float* __restrict__ out, int count4) {
  int i = blockIdx.x * blockDim.x + threadIdx.x;
  if (i >= count4) return;
  float4 a = ((const float4*)g0)[i];
  float4 b = ((const float4*)g1)[i];
  float4 o = { 2.0f * a.x + b.x, 2.0f * a.y + b.y,
               2.0f * a.z + b.z, 2.0f * a.w + b.w };
  ((float4*)out)[i] = o;
}

// ---------------------------------------------------------------------------
extern "C" void kernel_launch(void* const* d_in, const int* in_sizes, int n_in,
                              void* d_out, int out_size, void* d_ws, size_t ws_size,
                              hipStream_t stream) {
  const float* x  = (const float*)d_in[0];
  const int*   ei = (const int*)d_in[1];
  const int N = in_sizes[0] / 128;
  const int E = in_sizes[1] / 2;

  // cell params: [w_pre0, b_pre0, w_pre1, b_pre1, w_g0, b_g0, w_g1, b_g1]
  const float* P0[8]; const float* P1[8];
  for (int i = 0; i < 8; ++i) { P0[i] = (const float*)d_in[2 + i];
                                P1[i] = (const float*)d_in[10 + i]; }
  const float* w_cls = (const float*)d_in[18];
  const float* b_cls = (const float*)d_in[19];

  float* dinv = (float*)d_ws;
  size_t slab = (size_t)N * 128;
  float* S0 = dinv + N;
  float* S1 = S0 + slab;
  float* S2 = S1 + slab;

  const int TB = 256;
  const int gemmBlocks = (N + 127) / 128;
  auto gemm128 = [&](const float* X, const float* W, const float* B, float* Y) {
    gemm_bf16_wmma<128><<<gemmBlocks, 256, 0, stream>>>(X, W, B, Y, N);
  };
  auto gcn_agg = [&](const float* h2, const float* bias, float* out) {
    agg_init <<<(N * 32 + TB - 1) / TB, TB, 0, stream>>>(h2, dinv, bias, out, N);
    agg_edges<<<((size_t)E * 32 + TB - 1) / TB, TB, 0, stream>>>(h2, ei, dinv, out, E);
  };

  // symmetric GCN normalization (self-loops included)
  deg_init <<<(N + TB - 1) / TB, TB, 0, stream>>>(dinv, N);
  deg_edges<<<(E + TB - 1) / TB, TB, 0, stream>>>(ei, dinv, E);
  deg_rsqrt<<<(N + TB - 1) / TB, TB, 0, stream>>>(dinv, N);

  // cell(s0, s1, P) -> S2 ; needs only slabs S0..S2
  auto cell = [&](const float* s0, const float* s1, const float* const* P) {
    gemm128(s0, P[0], P[1], S0);          // t0 = s0@wp0 + bp0
    gemm128(s1, P[2], P[3], S1);          // t1 = s1@wp1 + bp1
    gemm128(S0, P[4], nullptr, S2);       // h0 = t0@wg0
    gcn_agg(S2, P[5], S0);                // g0 = Agg(h0) + bg0
    gemm128(S1, P[6], nullptr, S2);       // h1 = t1@wg1
    gcn_agg(S2, P[7], S1);                // g1 = Agg(h1) + bg1
    combine2<<<(N * 32 + TB - 1) / TB, TB, 0, stream>>>(S0, S1, S2, N * 32);
  };

  cell(x, x, P0);       // s1'  = cell(x, x)    -> S2
  cell(x, S2, P1);      // s1'' = cell(x, s1')  -> S2 (S2 read then reused)
  gemm_bf16_wmma<40><<<gemmBlocks, 256, 0, stream>>>(S2, w_cls, b_cls,
                                                     (float*)d_out, N);
}
